// DeepSetGraphClassifier_84490596647533
// MI455X (gfx1250) — matrix operand order repaired and checked by reading, not compiled, not run on hardware
//
#include <hip/hip_runtime.h>

typedef __attribute__((ext_vector_type(16))) _Float16 v16h;
typedef __attribute__((ext_vector_type(8)))  _Float16 v8h;
typedef __attribute__((ext_vector_type(8)))  float    v8f;

#define HDIM 128
#define TPB 256

// ---------------- elementwise / utility kernels ----------------

__global__ void k_fill_f32(float* p, float v, int n) {
    int i = blockIdx.x * blockDim.x + threadIdx.x;
    if (i < n) p[i] = v;
}

__global__ void k_deg_init(int* deg, int n) {          // self-loop contributes 1
    int i = blockIdx.x * blockDim.x + threadIdx.x;
    if (i < n) deg[i] = 1;
}

__global__ void k_deg_edges(const int* __restrict__ dst, int* deg, int e) {
    int i = blockIdx.x * blockDim.x + threadIdx.x;
    if (i < e) atomicAdd(&deg[dst[i]], 1);
}

__global__ void k_dinv(const int* __restrict__ deg, float* dinv, int n) {
    int i = blockIdx.x * blockDim.x + threadIdx.x;
    if (i < n) dinv[i] = rsqrtf((float)deg[i]);        // deg >= 1 always
}

__global__ void k_f32_to_f16(const float* __restrict__ s, _Float16* d, int n) {
    int i = blockIdx.x * blockDim.x + threadIdx.x;
    if (i < n) d[i] = (_Float16)s[i];
}

// WT[c*rows + r] = (f16) W[r*cols + c]   (rows == cols == 128 here)
__global__ void k_transpose_f16(const float* __restrict__ W, _Float16* WT, int rows, int cols) {
    int i = blockIdx.x * blockDim.x + threadIdx.x;
    if (i < rows * cols) {
        int r = i / cols, c = i % cols;
        WT[c * rows + r] = (_Float16)W[i];
    }
}

// ---------------- WMMA GEMM: C[M x 128] = A[M x 128] * W  (W given transposed, f16) ----

__global__ void k_wmma_gemm_h128(const _Float16* __restrict__ A,
                                 const _Float16* __restrict__ WT,
                                 float* __restrict__ C, int M) {
    const int tm = blockIdx.x;                 // 16-row tile; 8 waves cover 8 col tiles
    if (tm * 16 >= M) return;                  // wave-uniform: EXEC stays all-ones
    const int tn   = threadIdx.x >> 5;         // wave id 0..7 = column tile
    const int lane = threadIdx.x & 31;
    const int lr   = lane & 15;
    const int hh   = lane >> 4;

    const _Float16* arow = A  + (size_t)(tm * 16 + lr) * HDIM;  // A row m (row-major)
    const _Float16* brow = WT + (size_t)(tn * 16 + lr) * HDIM;  // W^T row n = column of W

    v8f c = {0.f, 0.f, 0.f, 0.f, 0.f, 0.f, 0.f, 0.f};
#pragma unroll
    for (int kk = 0; kk < HDIM; kk += 32) {
        // A lane layout: elems 0..7 -> K = kk + hh*8 + j ; elems 8..15 -> K = kk + 16 + hh*8 + j
        v8h a0 = *(const v8h*)(arow + kk + hh * 8);
        v8h a1 = *(const v8h*)(arow + kk + 16 + hh * 8);
        // B lane layout: elems 0..7 -> K = kk + hh*8 + j ; elems 8..15 -> K = kk + 16 + hh*8 + j of column n
        v8h b0 = *(const v8h*)(brow + kk + hh * 8);
        v8h b1 = *(const v8h*)(brow + kk + 16 + hh * 8);
        v16h a, b;
#pragma unroll
        for (int j = 0; j < 8; ++j) { a[j] = a0[j]; a[j + 8] = a1[j]; b[j] = b0[j]; b[j + 8] = b1[j]; }
        c = __builtin_amdgcn_wmma_f32_16x16x32_f16(false, a, false, b, (short)0, c, false, false);
    }
    // C lane layout: col = lr, rows = hh*8 + j
    float* crow = C + (size_t)(tm * 16 + hh * 8) * HDIM + tn * 16 + lr;
#pragma unroll
    for (int j = 0; j < 8; ++j) crow[(size_t)j * HDIM] = c[j];
}

// ---------------- GCN message passing ----------------

// self-loop term: acc[i][f] = hW[i][f] * dinv[i]^2
__global__ void k_init_acc(const float* __restrict__ hW, const float* __restrict__ dinv,
                           float* acc, int nelem) {
    int idx = blockIdx.x * blockDim.x + threadIdx.x;
    if (idx < nelem) {
        float di = dinv[idx >> 7];
        acc[idx] = hW[idx] * di * di;
    }
}

// edge scatter: 32 lanes per edge, 4 floats per lane
__global__ void k_scatter(const int* __restrict__ src, const int* __restrict__ dst,
                          const float* __restrict__ dinv, const float* __restrict__ hW,
                          float* acc, int E_) {
    int t = blockIdx.x * blockDim.x + threadIdx.x;
    int e = t >> 5;
    if (e >= E_) return;
    int f = (t & 31) * 4;
    int s = src[e], d = dst[e];
    float w = dinv[s] * dinv[d];
    const float4 v = *(const float4*)(hW + (size_t)s * HDIM + f);
    float* o = acc + (size_t)d * HDIM + f;
    atomicAdd(o + 0, v.x * w);
    atomicAdd(o + 1, v.y * w);
    atomicAdd(o + 2, v.z * w);
    atomicAdd(o + 3, v.w * w);
}

// h = relu(acc + b); write f32 (for pooling) and f16 (for next GEMM)
__global__ void k_bias_relu_dual(const float* __restrict__ acc, const float* __restrict__ b,
                                 float* h32, _Float16* h16, int nelem) {
    int idx = blockIdx.x * blockDim.x + threadIdx.x;
    if (idx < nelem) {
        float h = acc[idx] + b[idx & (HDIM - 1)];
        h = h > 0.f ? h : 0.f;
        h32[idx] = h;
        h16[idx] = (_Float16)h;
    }
}

// ---------------- pooling / DeepSets ----------------

__global__ void k_pool_accum(const float* __restrict__ h, const int* __restrict__ batch,
                             float* sums, float* cnts, int nelem) {
    int idx = blockIdx.x * blockDim.x + threadIdx.x;
    if (idx < nelem) {
        int i = idx >> 7, f = idx & (HDIM - 1);
        int g = batch[i];
        atomicAdd(sums + (size_t)g * HDIM + f, h[idx]);
        if (f == 0) atomicAdd(cnts + g, 1.f);
    }
}

__global__ void k_pool_finish(const float* __restrict__ sums, const float* __restrict__ cnts,
                              _Float16* emb16, int nelem) {
    int idx = blockIdx.x * blockDim.x + threadIdx.x;
    if (idx < nelem) {
        float cnt = cnts[idx >> 7];
        emb16[idx] = (_Float16)(sums[idx] / fmaxf(cnt, 1.f));
    }
}

__global__ void k_bias_relu_f16(const float* __restrict__ u, const float* __restrict__ b,
                                _Float16* out16, int nelem) {
    int idx = blockIdx.x * blockDim.x + threadIdx.x;
    if (idx < nelem) {
        float t = u[idx] + b[idx & (HDIM - 1)];
        out16[idx] = (_Float16)(t > 0.f ? t : 0.f);
    }
}

__global__ void k_bias_tanh(const float* __restrict__ v, const float* __restrict__ b,
                            float* p, int nelem) {
    int idx = blockIdx.x * blockDim.x + threadIdx.x;
    if (idx < nelem) p[idx] = tanhf(v[idx] + b[idx & (HDIM - 1)]);
}

__global__ void k_agg(const float* __restrict__ p, const int* __restrict__ set_batch,
                      float* agg, int nelem) {
    int idx = blockIdx.x * blockDim.x + threadIdx.x;
    if (idx < nelem) {
        int g = idx >> 7, f = idx & (HDIM - 1);
        atomicAdd(agg + (size_t)set_batch[g] * HDIM + f, p[idx]);
    }
}

// a1[s][j] = relu(dot(agg[s], phiW1[:,j]) + phib1[j])   (S*128 threads)
__global__ void k_final1(const float* __restrict__ agg, const float* __restrict__ W,
                         const float* __restrict__ b, float* a1, int nelem) {
    int idx = blockIdx.x * blockDim.x + threadIdx.x;
    if (idx < nelem) {
        int s = idx >> 7, j = idx & (HDIM - 1);
        float acc = b[j];
        const float* ar = agg + (size_t)s * HDIM;
#pragma unroll 8
        for (int k = 0; k < HDIM; ++k) acc += ar[k] * W[k * HDIM + j];
        a1[idx] = acc > 0.f ? acc : 0.f;
    }
}

// out[s][c] = dot(a1[s], phiW2[:,c]) + phib2[c]   (S*C threads, C=10)
__global__ void k_final2(const float* __restrict__ a1, const float* __restrict__ W,
                         const float* __restrict__ b, float* out, int nelem) {
    int idx = blockIdx.x * blockDim.x + threadIdx.x;
    if (idx < nelem) {
        int s = idx / 10, c = idx % 10;
        float acc = b[c];
        const float* ar = a1 + (size_t)s * HDIM;
#pragma unroll 8
        for (int k = 0; k < HDIM; ++k) acc += ar[k] * W[k * 10 + c];
        out[idx] = acc;
    }
}

// ---------------- host launcher ----------------

static inline int cdiv(long long a, int b) { return (int)((a + b - 1) / b); }

extern "C" void kernel_launch(void* const* d_in, const int* in_sizes, int n_in,
                              void* d_out, int out_size, void* d_ws, size_t ws_size,
                              hipStream_t stream) {
    const float* x        = (const float*)d_in[0];
    const int*   eidx     = (const int*)d_in[1];
    const int*   batch    = (const int*)d_in[2];
    const int*   setb     = (const int*)d_in[3];
    const float* Wg[3]    = { (const float*)d_in[4], (const float*)d_in[6], (const float*)d_in[8] };
    const float* bg[3]    = { (const float*)d_in[5], (const float*)d_in[7], (const float*)d_in[9] };
    const float* psiW1    = (const float*)d_in[10];
    const float* psib1    = (const float*)d_in[11];
    const float* psiW2    = (const float*)d_in[12];
    const float* psib2    = (const float*)d_in[13];
    const float* phiW1    = (const float*)d_in[14];
    const float* phib1    = (const float*)d_in[15];
    const float* phiW2    = (const float*)d_in[16];
    const float* phib2    = (const float*)d_in[17];
    float* out = (float*)d_out;

    const int N = in_sizes[0] / HDIM;
    const int E = in_sizes[1] / 2;
    const int G = in_sizes[3];
    const int S = out_size / 10;
    const int src_off = 0, dst_off = E;
    const int NE = N * HDIM;   // 12.8M
    const int GE = G * HDIM;
    const int SE = S * HDIM;

    // workspace carve-out (256-B aligned)
    char* wp = (char*)d_ws;
    auto carve = [&](size_t bytes) -> char* {
        char* p = wp;
        wp += (bytes + 255) & ~(size_t)255;
        return p;
    };
    _Float16* buf0  = (_Float16*)carve((size_t)NE * 2);  // f16 activations (GEMM A)
    float*    buf1  = (float*)   carve((size_t)NE * 4);  // GEMM out / final h f32
    float*    buf2  = (float*)   carve((size_t)NE * 4);  // scatter accumulator
    float*    dinv  = (float*)   carve((size_t)N * 4);
    int*      deg   = (int*)     carve((size_t)N * 4);
    _Float16* wt[5];
    for (int i = 0; i < 5; ++i) wt[i] = (_Float16*)carve((size_t)HDIM * HDIM * 2);
    float*    sums  = (float*)   carve((size_t)GE * 4);
    float*    cnts  = (float*)   carve((size_t)G * 4);
    _Float16* emb16 = (_Float16*)carve((size_t)GE * 2);
    float*    ubuf  = (float*)   carve((size_t)GE * 4);
    _Float16* t1f16 = (_Float16*)carve((size_t)GE * 2);
    float*    pbuf  = (float*)   carve((size_t)GE * 4);
    float*    agg   = (float*)   carve((size_t)SE * 4);
    float*    a1    = (float*)   carve((size_t)SE * 4);
    (void)ws_size; (void)n_in;

    // degree / normalization
    k_deg_init <<<cdiv(N, TPB), TPB, 0, stream>>>(deg, N);
    k_deg_edges<<<cdiv(E, TPB), TPB, 0, stream>>>(eidx + dst_off, deg, E);
    k_dinv     <<<cdiv(N, TPB), TPB, 0, stream>>>(deg, dinv, N);

    // input + weight conversion to f16 (weights transposed for B-fragments)
    k_f32_to_f16<<<cdiv(NE, TPB), TPB, 0, stream>>>(x, buf0, NE);
    const float* wsrc[5] = { Wg[0], Wg[1], Wg[2], psiW1, psiW2 };
    for (int i = 0; i < 5; ++i)
        k_transpose_f16<<<cdiv(HDIM * HDIM, TPB), TPB, 0, stream>>>(wsrc[i], wt[i], HDIM, HDIM);

    // three GCN layers
    for (int l = 0; l < 3; ++l) {
        k_wmma_gemm_h128<<<N / 16, TPB, 0, stream>>>(buf0, wt[l], buf1, N);
        k_init_acc      <<<cdiv(NE, TPB), TPB, 0, stream>>>(buf1, dinv, buf2, NE);
        k_scatter       <<<cdiv((long long)E * 32, TPB), TPB, 0, stream>>>(
                            eidx + src_off, eidx + dst_off, dinv, buf1, buf2, E);
        k_bias_relu_dual<<<cdiv(NE, TPB), TPB, 0, stream>>>(buf2, bg[l], buf1, buf0, NE);
    }

    // mean pool per graph
    k_fill_f32  <<<cdiv(GE, TPB), TPB, 0, stream>>>(sums, 0.f, GE);
    k_fill_f32  <<<cdiv(G,  TPB), TPB, 0, stream>>>(cnts, 0.f, G);
    k_pool_accum<<<cdiv(NE, TPB), TPB, 0, stream>>>(buf1, batch, sums, cnts, NE);
    k_pool_finish<<<cdiv(GE, TPB), TPB, 0, stream>>>(sums, cnts, emb16, GE);

    // DeepSets psi: tanh(relu(emb @ psiW1 + b) @ psiW2 + b)
    k_wmma_gemm_h128<<<G / 16, TPB, 0, stream>>>(emb16, wt[3], ubuf, G);
    k_bias_relu_f16 <<<cdiv(GE, TPB), TPB, 0, stream>>>(ubuf, psib1, t1f16, GE);
    k_wmma_gemm_h128<<<G / 16, TPB, 0, stream>>>(t1f16, wt[4], ubuf, G);
    k_bias_tanh     <<<cdiv(GE, TPB), TPB, 0, stream>>>(ubuf, psib2, pbuf, GE);

    // segment-sum over sets, then phi head
    k_fill_f32<<<cdiv(SE, TPB), TPB, 0, stream>>>(agg, 0.f, SE);
    k_agg     <<<cdiv(GE, TPB), TPB, 0, stream>>>(pbuf, setb, agg, GE);
    k_final1  <<<cdiv(SE, TPB), TPB, 0, stream>>>(agg, phiW1, phib1, a1, SE);
    k_final2  <<<cdiv(S * 10, TPB), TPB, 0, stream>>>(a1, phiW2, phib2, out, S * 10);
}
